// Structured_Attention_1709396983949
// MI455X (gfx1250) — matrix operand order repaired and checked by reading, not compiled, not run on hardware
//
#include <hip/hip_runtime.h>
#include <hip/hip_bf16.h>

#define B_  16
#define N_  2048
#define SS  2048   // SIZE_S
#define SH  512    // SIZE_H
#define DI  512    // D_INT

typedef __attribute__((ext_vector_type(16))) __bf16 v16bf;
typedef __attribute__((ext_vector_type(8)))  float  v8f;

__device__ __forceinline__ unsigned short f2bf(float f) {
    unsigned int u = __float_as_uint(f);
    u += 0x7FFFu + ((u >> 16) & 1u);           // round-to-nearest-even
    return (unsigned short)(u >> 16);
}
__device__ __forceinline__ unsigned int pack2(float a, float b) {
    return (unsigned int)f2bf(a) | ((unsigned int)f2bf(b) << 16);
}

// ---------------- W1 (f32 [DI][SS]) -> bf16, same layout ----------------
__global__ void cvt_w1_kernel(const float* __restrict__ W1,
                              unsigned short* __restrict__ Wb) {
    int idx = (blockIdx.x * 256 + threadIdx.x) * 4;
    float4 f = *(const float4*)(W1 + idx);
    uint2 o;
    o.x = pack2(f.x, f.y);
    o.y = pack2(f.z, f.w);
    *(uint2*)(Wb + idx) = o;
}

// ---------------- g[b,i] = b1[i] + b2[i] + sum_h h[b,h]*W2[i,h] ----------------
__global__ void g_kernel(const float* __restrict__ h, const float* __restrict__ W2,
                         const float* __restrict__ b1, const float* __restrict__ b2,
                         float* __restrict__ g) {
    int b = blockIdx.x;
    const float* hr = h + b * SH;
    for (int rep = 0; rep < 2; ++rep) {
        int i = threadIdx.x + rep * 256;
        float acc = b1[i] + b2[i];
        const float* wr = W2 + (size_t)i * SH;
        for (int k = 0; k < SH; k += 4) {
            float4 hv = *(const float4*)(hr + k);
            float4 wv = *(const float4*)(wr + k);
            acc += hv.x * wv.x + hv.y * wv.y + hv.z * wv.z + hv.w * wv.w;
        }
        g[b * DI + i] = acc;
    }
}

// ---------------- scores[b,n] = sum_i relu(S.W1^T + g) * W3 + b3 ----------------
// block: 256 threads = 8 waves. Block covers (batch b, 32 rows of n, all 512 i).
// Wave w owns i-columns [w*64, w*64+64) as 4 WMMA tiles of 16; each B fragment
// feeds two A row-tiles, and the next i-tile's B fragment is prefetched while
// the current one is consumed (keeps loads outstanding across the wmmas).
__global__ void __launch_bounds__(256) scores_kernel(
    const float* __restrict__ Sg, const unsigned short* __restrict__ Wb,
    const float* __restrict__ g,  const float* __restrict__ W3,
    const float* __restrict__ b3, float* __restrict__ scores)
{
    __shared__ __align__(16) unsigned int As[2][32 * 32]; // 2 x (32 rows x 64 bf16)
    __shared__ float sc[32];

    const int n0   = blockIdx.x * 32;
    const int b    = blockIdx.y;
    const int tid  = threadIdx.x;
    const int lane = tid & 31;
    const int wave = tid >> 5;
    const int m    = lane & 15;
    const int hi   = lane >> 4;

    if (tid < 32) sc[tid] = 0.0f;

    const float* sBase = Sg + ((size_t)b * N_ + n0) * SS;

    union AB { v16bf v; uint4 q[2]; };
    v8f acc[2][4] = {};

    int p = 0;
    for (int k0 = 0; k0 < SS; k0 += 64, p ^= 1) {
        // stage 32x64 f32 tile of S -> packed bf16 pairs (1024 words, 512 quads)
        #pragma unroll
        for (int it = 0; it < 2; ++it) {
            int q = tid * 2 + it;
            int row = q >> 4, qc = q & 15;
            float4 f = *(const float4*)(sBase + (size_t)row * SS + k0 + qc * 4);
            As[p][row * 32 + qc * 2 + 0] = pack2(f.x, f.y);
            As[p][row * 32 + qc * 2 + 1] = pack2(f.z, f.w);
        }
        __syncthreads();   // single barrier/iter: double buffer covers WAR hazard

        // A fragments (contiguous k = 16*hi + s, matching B) for the 2 row-tiles
        AB a[2][2];
        #pragma unroll
        for (int rt = 0; rt < 2; ++rt) {
            const unsigned int* ap = &As[p][(rt * 16 + m) * 32 + hi * 8];
            a[rt][0].q[0] = *(const uint4*)(ap);
            a[rt][0].q[1] = *(const uint4*)(ap + 4);
            a[rt][1].q[0] = *(const uint4*)(ap + 16);
            a[rt][1].q[1] = *(const uint4*)(ap + 20);
        }

        // B fragments (contiguous k order, same as A: lanes 0..15 K0..15,
        // lanes 16..31 K16..31), software pipelined across the 4 i-tiles
        const unsigned short* wbase =
            Wb + (size_t)(wave * 64 + m) * SS + k0 + hi * 16;
        AB b0c, b1c, b0n, b1n;
        b0c.q[0] = *(const uint4*)(wbase);
        b0c.q[1] = *(const uint4*)(wbase + 8);
        b1c.q[0] = *(const uint4*)(wbase + 32);
        b1c.q[1] = *(const uint4*)(wbase + 40);
        #pragma unroll
        for (int t = 0; t < 4; ++t) {
            if (t < 3) {
                const unsigned short* wn = wbase + (size_t)(t + 1) * 16 * SS;
                b0n.q[0] = *(const uint4*)(wn);
                b0n.q[1] = *(const uint4*)(wn + 8);
                b1n.q[0] = *(const uint4*)(wn + 32);
                b1n.q[1] = *(const uint4*)(wn + 40);
            }
            #pragma unroll
            for (int rt = 0; rt < 2; ++rt) {
                acc[rt][t] = __builtin_amdgcn_wmma_f32_16x16x32_bf16(
                    false, a[rt][0].v, false, b0c.v, (short)0, acc[rt][t], false, false);
                acc[rt][t] = __builtin_amdgcn_wmma_f32_16x16x32_bf16(
                    false, a[rt][1].v, false, b1c.v, (short)0, acc[rt][t], false, false);
            }
            b0c = b0n;
            b1c = b1n;
        }
    }

    // epilogue: fused = relu(fc1 + g); score += fused * W3
    // C layout: VGPR r, lane l -> n_local = rt*16 + r + 8*hi, i = tile*16 + (l%16)
    const float* gr = g + b * DI;
    #pragma unroll
    for (int t = 0; t < 4; ++t) {
        int i = wave * 64 + t * 16 + m;
        float gi = gr[i];
        float w3 = W3[i];
        #pragma unroll
        for (int rt = 0; rt < 2; ++rt) {
            #pragma unroll
            for (int r = 0; r < 8; ++r) {
                float v = acc[rt][t][r] + gi;
                v = fmaxf(v, 0.0f) * w3;
                v += __shfl_xor(v, 1, 16);
                v += __shfl_xor(v, 2, 16);
                v += __shfl_xor(v, 4, 16);
                v += __shfl_xor(v, 8, 16);
                if (m == 0) atomicAdd(&sc[rt * 16 + r + hi * 8], v);
            }
        }
    }
    __syncthreads();
    if (tid < 32) scores[(size_t)b * N_ + n0 + tid] = sc[tid] + b3[0];
}

// ---------------- softmax over N per batch ----------------
__global__ void softmax_kernel(const float* __restrict__ scores,
                               float* __restrict__ a_t) {
    __shared__ float red[256];
    const int b = blockIdx.x, tid = threadIdx.x;
    const float* sr = scores + b * N_;
    float x[8];
    float lmax = -3.402823466e38f;
    #pragma unroll
    for (int j = 0; j < 8; ++j) { x[j] = sr[tid + j * 256]; lmax = fmaxf(lmax, x[j]); }
    red[tid] = lmax; __syncthreads();
    for (int s = 128; s > 0; s >>= 1) {
        if (tid < s) red[tid] = fmaxf(red[tid], red[tid + s]);
        __syncthreads();
    }
    float mx = red[0]; __syncthreads();
    float lsum = 0.0f;
    #pragma unroll
    for (int j = 0; j < 8; ++j) { x[j] = __expf(x[j] - mx); lsum += x[j]; }
    red[tid] = lsum; __syncthreads();
    for (int s = 128; s > 0; s >>= 1) {
        if (tid < s) red[tid] += red[tid + s];
        __syncthreads();
    }
    float inv = 1.0f / red[0];
    #pragma unroll
    for (int j = 0; j < 8; ++j) a_t[b * N_ + tid + j * 256] = x[j] * inv;
}

// ---------------- z_t[b,s] = sum_n a_t[b,n] * S[b,n,s] (bandwidth bound) ----------------
__global__ void __launch_bounds__(256) zt_kernel(const float* __restrict__ Sg,
                                                 const float* __restrict__ a_t,
                                                 float* __restrict__ z_t) {
    __shared__ float aLds[N_];
    const int b   = blockIdx.y;
    const int c0  = blockIdx.x * 1024;
    const int tid = threadIdx.x;
    for (int j = tid; j < N_; j += 256) aLds[j] = a_t[b * N_ + j];
    __syncthreads();
    const float* sp = Sg + (size_t)b * N_ * SS + c0 + tid * 4;
    float4 acc = {0.0f, 0.0f, 0.0f, 0.0f};
    #pragma unroll 4
    for (int n = 0; n < N_; ++n) {
        float a = aLds[n];
        float4 sv = *(const float4*)(sp + (size_t)n * SS);
        acc.x += a * sv.x; acc.y += a * sv.y;
        acc.z += a * sv.z; acc.w += a * sv.w;
    }
    *(float4*)(z_t + b * SS + c0 + tid * 4) = acc;
}

extern "C" void kernel_launch(void* const* d_in, const int* in_sizes, int n_in,
                              void* d_out, int out_size, void* d_ws, size_t ws_size,
                              hipStream_t stream) {
    const float* Sg = (const float*)d_in[0];
    const float* h  = (const float*)d_in[1];
    const float* W1 = (const float*)d_in[2];
    const float* b1 = (const float*)d_in[3];
    const float* W2 = (const float*)d_in[4];
    const float* b2 = (const float*)d_in[5];
    const float* W3 = (const float*)d_in[6];
    const float* b3 = (const float*)d_in[7];

    float* z_t = (float*)d_out;                 // [B, SIZE_S]
    float* a_t = z_t + (size_t)B_ * SS;         // [B, N] (tuple order: z_t, a_t)

    char* ws = (char*)d_ws;
    unsigned short* Wb = (unsigned short*)ws;                       // 2 MB bf16 W1
    float* g      = (float*)(ws + 2u * 1024 * 1024);                // 32 KB
    float* scores = (float*)(ws + 2u * 1024 * 1024 + 32u * 1024);   // 128 KB

    cvt_w1_kernel<<<1024, 256, 0, stream>>>(W1, Wb);                    // 512*2048 elems
    g_kernel<<<B_, 256, 0, stream>>>(h, W2, b1, b2, g);
    scores_kernel<<<dim3(N_ / 32, B_), 256, 0, stream>>>(Sg, Wb, g, W3, b3, scores);
    softmax_kernel<<<B_, 256, 0, stream>>>(scores, a_t);
    zt_kernel<<<dim3(SS / 1024, B_), 256, 0, stream>>>(Sg, a_t, z_t);
}
